// GAT_DANN_12214886990281
// MI455X (gfx1250) — matrix operand order repaired and checked by reading, not compiled, not run on hardware
//
#include <hip/hip_runtime.h>
#include <hip/hip_bf16.h>
#include <math.h>

// ---------------------------------------------------------------------------
// Types for CDNA5 WMMA
// ---------------------------------------------------------------------------
typedef __bf16 bf16_t;
typedef __attribute__((ext_vector_type(2)))  __bf16 v2bf;
typedef __attribute__((ext_vector_type(16))) __bf16 v16bf;
typedef __attribute__((ext_vector_type(8)))  float  v8f;

#define NEG_SLOPE 0.2f
#define BN_EPS    1e-5f
#define NGROUP    64
#define NCLS      10
#define FOUT      128

static __device__ __forceinline__ float leaky(float v) {
    return v > 0.0f ? v : NEG_SLOPE * v;
}

// monotone float <-> uint key (total order preserved under unsigned compare)
static __device__ __forceinline__ unsigned fkey(float f) {
    unsigned u = __float_as_uint(f);
    return (u & 0x80000000u) ? ~u : (u | 0x80000000u);
}
static __device__ __forceinline__ float fdec(unsigned u) {
    return (u & 0x80000000u) ? __uint_as_float(u ^ 0x80000000u)
                             : __uint_as_float(~u);
}

static __device__ __forceinline__ void edge_sd(const int* __restrict__ ei,
                                               int E, int e, int& s, int& d) {
    if (e < E) { s = ei[e]; d = ei[E + e]; }
    else       { s = e - E; d = s; }       // appended self-loop
}

// ---------------------------------------------------------------------------
// WMMA GEMM:  C[M,Ncols] = A[M,K] @ B[K,Ncols]   (f32 mem, bf16 compute,
// f32 accum).  Block = 128 threads = 4 waves, 64x64 tile, K-step 32.
// LDS tiles in *fragment order* (lane-contiguous 32B), double buffered;
// next tile is prefetched into registers while WMMAs run on current tile.
// ---------------------------------------------------------------------------

// gather next tile's globals into registers (branch-free when !GUARD)
template <bool GUARD>
static __device__ __forceinline__ void gload_tile(
    const float* __restrict__ A, const float* __restrict__ B,
    int M, int K, int Ncols, int rowBlock, int colBlock, int k0, int tid,
    float2 aV[8], float bV0[8], float bV1[8]) {
    #pragma unroll
    for (int j = 0; j < 8; ++j) {
        int i = tid + j * 128;
        int r = i >> 4, k = (i & 15) << 1;   // 64 rows x 16 k-pairs
        int gr = rowBlock + r;
        if (!GUARD || gr < M) {
            aV[j] = *(const float2*)&A[(size_t)gr * K + (k0 + k)];
        } else {
            aV[j].x = 0.0f; aV[j].y = 0.0f;
        }
    }
    #pragma unroll
    for (int j = 0; j < 8; ++j) {
        int i = tid + j * 128;
        int k = (i >> 6) << 1, n = i & 63;   // 16 k-pairs x 64 cols
        const float* bp = &B[(size_t)(k0 + k) * Ncols + (colBlock + n)];
        bV0[j] = bp[0];
        bV1[j] = bp[Ncols];
    }
}

// convert + scatter registers into fragment-order LDS
static __device__ __forceinline__ void store_tile(
    bf16_t* __restrict__ AsF, bf16_t* __restrict__ BsF, int tid,
    const float2 aV[8], const float bV0[8], const float bV1[8]) {
    #pragma unroll
    for (int j = 0; j < 8; ++j) {
        int i = tid + j * 128;
        int r = i >> 4, k = (i & 15) << 1;
        // A-frag layout (16-bit A 16x32): lane=(r&15)+16*((k&15)>=8),
        // e = 8*(k>=16)+(k&7); pair (k,k+1) -> (e,e+1) same lane
        int wv = r >> 4;
        int ln = (r & 15) + (((k & 15) >= 8) ? 16 : 0);
        int e  = ((k >> 4) << 3) + (k & 7);
        v2bf p; p[0] = (bf16_t)aV[j].x; p[1] = (bf16_t)aV[j].y;
        *(v2bf*)&AsF[(((wv << 5) + ln) << 4) + e] = p;
    }
    #pragma unroll
    for (int j = 0; j < 8; ++j) {
        int i = tid + j * 128;
        int k = (i >> 6) << 1, n = i & 63;
        // B-frag layout (dense 32x16): lane=(n&15)+16*((k&15)>=8), e=8*(k>=16)+(k&7)
        int ct = n >> 4;
        int ln = (n & 15) + (((k & 15) >= 8) ? 16 : 0);
        int e  = ((k >> 4) << 3) + (k & 7);
        v2bf p; p[0] = (bf16_t)bV0[j]; p[1] = (bf16_t)bV1[j];
        *(v2bf*)&BsF[(((ct << 5) + ln) << 4) + e] = p;
    }
}

template <bool GUARD>
static __device__ __forceinline__ void gemm_body(
    const float* __restrict__ A, const float* __restrict__ B,
    float* __restrict__ C, int M, int K, int Ncols,
    bf16_t* __restrict__ AsF, bf16_t* __restrict__ BsF) {
    const int tid  = threadIdx.x;
    const int lane = tid & 31;
    const int wave = tid >> 5;
    const int rowBlock = blockIdx.x * 64;
    const int colBlock = blockIdx.y * 64;

    v8f acc[4];
    #pragma unroll
    for (int c = 0; c < 4; ++c)
        #pragma unroll
        for (int j = 0; j < 8; ++j) acc[c][j] = 0.0f;

    float2 aV[8]; float bV0[8], bV1[8];
    gload_tile<GUARD>(A, B, M, K, Ncols, rowBlock, colBlock, 0, tid, aV, bV0, bV1);

    int buf = 0;
    for (int k0 = 0; k0 < K; k0 += 32) {
        store_tile(AsF + buf * (64 * 32), BsF + buf * (64 * 32), tid, aV, bV0, bV1);
        __syncthreads();

        // prefetch next tile while WMMAs run
        if (k0 + 32 < K)
            gload_tile<GUARD>(A, B, M, K, Ncols, rowBlock, colBlock, k0 + 32,
                              tid, aV, bV0, bV1);

        const bf16_t* ab = AsF + buf * (64 * 32);
        const bf16_t* bb = BsF + buf * (64 * 32);
        const v16bf afrag = *(const v16bf*)&ab[((wave << 5) + lane) << 4];
        v16bf bfr[4];
        #pragma unroll
        for (int c = 0; c < 4; ++c)
            bfr[c] = *(const v16bf*)&bb[((c << 5) + lane) << 4];
        #pragma unroll
        for (int c = 0; c < 4; ++c)
            acc[c] = __builtin_amdgcn_wmma_f32_16x16x32_bf16(
                false, afrag, false, bfr[c], (short)0, acc[c], false, false);
        buf ^= 1;
    }

    // C/D layout: VGPR r -> M = r (+8 for lanes 16..31), N = lane&15
    const int mbase = rowBlock + (wave << 4) + ((lane >= 16) ? 8 : 0);
    const int nlane = lane & 15;
    #pragma unroll
    for (int c = 0; c < 4; ++c) {
        int col = colBlock + (c << 4) + nlane;
        #pragma unroll
        for (int r = 0; r < 8; ++r) {
            int row = mbase + r;
            if (!GUARD || row < M) C[(size_t)row * Ncols + col] = acc[c][r];
        }
    }
}

__global__ __launch_bounds__(128)
void gemm_wmma_bf16(const float* __restrict__ A, const float* __restrict__ B,
                    float* __restrict__ C, int M, int K, int Ncols) {
    __shared__ bf16_t AsF[2 * 64 * 32];
    __shared__ bf16_t BsF[2 * 64 * 32];
    if ((int)blockIdx.x * 64 + 64 <= M)
        gemm_body<false>(A, B, C, M, K, Ncols, AsF, BsF);
    else
        gemm_body<true>(A, B, C, M, K, Ncols, AsF, BsF);
}

// ---------------------------------------------------------------------------
// es[n,h] = sum_d h[n,h,d]*a_s[h,d] ; ed likewise.  Block = H*D threads.
// ---------------------------------------------------------------------------
__global__ void attn_coef(const float* __restrict__ h,
                          const float* __restrict__ a_s,
                          const float* __restrict__ a_d,
                          float* __restrict__ es, float* __restrict__ ed,
                          int H, int D) {
    __shared__ float svs[256], svd[256];
    const int n = blockIdx.x, t = threadIdx.x, HD = H * D;
    float p = h[(size_t)n * HD + t];
    svs[t] = p * a_s[t];
    svd[t] = p * a_d[t];
    __syncthreads();
    for (int s = D >> 1; s > 0; s >>= 1) {
        if ((t & (D - 1)) < s) { svs[t] += svs[t + s]; svd[t] += svd[t + s]; }
        __syncthreads();
    }
    if ((t & (D - 1)) == 0) {
        int hh = t / D;
        es[n * H + hh] = svs[t];
        ed[n * H + hh] = svd[t];
    }
}

// zero out-accumulator, z; set m to key(-inf)
__global__ void layer_init(float* __restrict__ obuf, unsigned* __restrict__ m,
                           float* __restrict__ z, int nHD, int nH) {
    int i = blockIdx.x * blockDim.x + threadIdx.x;
    if (i < nHD) obuf[i] = 0.0f;
    if (i < nH) { z[i] = 0.0f; m[i] = fkey(-INFINITY); }
}

__global__ void edge_max(const int* __restrict__ ei, int E, int Ep, int H,
                         const float* __restrict__ es, const float* __restrict__ ed,
                         unsigned* __restrict__ m) {
    int i = blockIdx.x * blockDim.x + threadIdx.x;
    if (i >= Ep * H) return;
    int e = i / H, hh = i - e * H, s, d;
    edge_sd(ei, E, e, s, d);
    float v = leaky(es[s * H + hh] + ed[d * H + hh]);
    atomicMax(&m[d * H + hh], fkey(v));
}

__global__ void edge_sum(const int* __restrict__ ei, int E, int Ep, int H,
                         const float* __restrict__ es, const float* __restrict__ ed,
                         const unsigned* __restrict__ m, float* __restrict__ z) {
    int i = blockIdx.x * blockDim.x + threadIdx.x;
    if (i >= Ep * H) return;
    int e = i / H, hh = i - e * H, s, d;
    edge_sd(ei, E, e, s, d);
    float v = leaky(es[s * H + hh] + ed[d * H + hh]);
    atomicAdd(&z[d * H + hh], __expf(v - fdec(m[d * H + hh])));
}

// one block per edge, H*D threads: out[dst,t] += h[src,t] * att[head(t)]
__global__ void aggregate(const int* __restrict__ ei, int E, int H, int D,
                          const float* __restrict__ es, const float* __restrict__ ed,
                          const unsigned* __restrict__ m, const float* __restrict__ z,
                          const float* __restrict__ h, float* __restrict__ obuf) {
    const int e = blockIdx.x, t = threadIdx.x, HD = H * D;
    int s, d;
    edge_sd(ei, E, e, s, d);
    const int hh = t / D;
    float v = leaky(es[s * H + hh] + ed[d * H + hh]);
    float att = __expf(v - fdec(m[d * H + hh])) / z[d * H + hh];
    atomicAdd(&obuf[(size_t)d * HD + t], h[(size_t)s * HD + t] * att);
}

// one block per column: mu, 1/sqrt(var+eps) over N rows (biased var)
__global__ __launch_bounds__(256)
void bn_stats(const float* __restrict__ obuf, int N, int HD,
              float* __restrict__ mu, float* __restrict__ inv) {
    __shared__ float ss[256], s2[256];
    const int col = blockIdx.x, t = threadIdx.x;
    float a = 0.0f, b = 0.0f;
    for (int r = t; r < N; r += 256) {
        float v = obuf[(size_t)r * HD + col];
        a += v; b += v * v;
    }
    ss[t] = a; s2[t] = b;
    __syncthreads();
    for (int s = 128; s > 0; s >>= 1) {
        if (t < s) { ss[t] += ss[t + s]; s2[t] += s2[t + s]; }
        __syncthreads();
    }
    if (t == 0) {
        float mean = ss[0] / (float)N;
        float var  = s2[0] / (float)N - mean * mean;
        mu[col]  = mean;
        inv[col] = rsqrtf(var + BN_EPS);
    }
}

// (note: the GAT bias b cancels inside BN, so it is skipped entirely)
__global__ void bn_apply_relu(const float* __restrict__ obuf,
                              const float* __restrict__ mu, const float* __restrict__ inv,
                              const float* __restrict__ gamma, const float* __restrict__ beta,
                              float* __restrict__ xout, int total, int HD) {
    int i = blockIdx.x * blockDim.x + threadIdx.x;
    if (i >= total) return;
    int col = i % HD;
    float y = (obuf[i] - mu[col]) * inv[col] * gamma[col] + beta[col];
    xout[i] = fmaxf(y, 0.0f);
}

// ---------------------------------------------------------------------------
// Pooling + heads
// ---------------------------------------------------------------------------
__global__ void pool_init(float* __restrict__ fsum, float* __restrict__ cnt) {
    int i = blockIdx.x * blockDim.x + threadIdx.x;
    if (i < NGROUP * FOUT) fsum[i] = 0.0f;
    if (i < NGROUP) cnt[i] = 0.0f;
}

__global__ void pool_accum(const float* __restrict__ x, const int* __restrict__ batch,
                           float* __restrict__ fsum, float* __restrict__ cnt, int N) {
    int i = blockIdx.x * blockDim.x + threadIdx.x;
    if (i >= N * FOUT) return;
    int n = i >> 7, c = i & (FOUT - 1);
    int g = batch[n];
    atomicAdd(&fsum[g * FOUT + c], x[i]);
    if (c == 0) atomicAdd(&cnt[g], 1.0f);
}

__global__ __launch_bounds__(128)
void heads_kernel(const float* __restrict__ fsum, const float* __restrict__ cnt,
                  const float* __restrict__ clfW, const float* __restrict__ clfb,
                  const float* __restrict__ dW1, const float* __restrict__ db1,
                  const float* __restrict__ dW2, const float* __restrict__ db2,
                  float* __restrict__ out) {
    __shared__ float feat[FOUT];
    __shared__ float dh[64];
    const int g = blockIdx.x, t = threadIdx.x;
    float c = fmaxf(cnt[g], 1.0f);
    feat[t] = fsum[g * FOUT + t] / c;
    __syncthreads();

    float* out_cls  = out;                               // [G, 10]
    float* out_dom  = out + NGROUP * NCLS;               // [G, 2]
    float* out_feat = out + NGROUP * NCLS + NGROUP * 2;  // [G, 128]

    out_feat[g * FOUT + t] = feat[t];

    if (t < NCLS) {
        float s = clfb[t];
        #pragma unroll 4
        for (int k = 0; k < FOUT; ++k) s += feat[k] * clfW[k * NCLS + t];
        out_cls[g * NCLS + t] = s;
    }
    if (t < 64) {
        float s = db1[t];
        #pragma unroll 4
        for (int k = 0; k < FOUT; ++k) s += feat[k] * dW1[k * 64 + t];
        dh[t] = fmaxf(s, 0.0f);
    }
    __syncthreads();
    if (t < 2) {
        float s = db2[t];
        #pragma unroll 4
        for (int k = 0; k < 64; ++k) s += dh[k] * dW2[k * 2 + t];
        out_dom[g * 2 + t] = s;
    }
}

// ---------------------------------------------------------------------------
// Launcher
// ---------------------------------------------------------------------------
extern "C" void kernel_launch(void* const* d_in, const int* in_sizes, int n_in,
                              void* d_out, int out_size, void* d_ws, size_t ws_size,
                              hipStream_t stream) {
    const float* x0    = (const float*)d_in[0];
    const int*   ei    = (const int*)d_in[1];
    const int*   batch = (const int*)d_in[2];
    const int N  = in_sizes[2];
    const int E  = in_sizes[1] / 2;
    const int Ep = E + N;

    // workspace carve-up (floats)
    float* ws   = (float*)d_ws;
    float* xbuf = ws;                        // N*256
    float* hbuf = xbuf + (size_t)N * 256;    // N*256
    float* obuf = hbuf + (size_t)N * 256;    // N*256
    float* es   = obuf + (size_t)N * 256;    // N*4
    float* ed   = es   + (size_t)N * 4;      // N*4
    unsigned* mbuf = (unsigned*)(ed + (size_t)N * 4);        // N*4
    float* zbuf = (float*)mbuf + (size_t)N * 4;              // N*4
    float* mu   = zbuf + (size_t)N * 4;      // 256
    float* inv  = mu + 256;                  // 256
    float* fsum = inv + 256;                 // 64*128
    float* cnt  = fsum + NGROUP * FOUT;      // 64

    const int Ks[3]   = {128, 256, 256};
    const int Hs[3]   = {4, 4, 1};
    const int Ds[3]   = {64, 64, 128};
    const int base[3] = {3, 9, 15};

    const float* curx = x0;
    for (int l = 0; l < 3; ++l) {
        const float* W     = (const float*)d_in[base[l] + 0];
        const float* a_s   = (const float*)d_in[base[l] + 1];
        const float* a_d   = (const float*)d_in[base[l] + 2];
        const float* gamma = (const float*)d_in[base[l] + 4];
        const float* beta  = (const float*)d_in[base[l] + 5];
        const int K = Ks[l], H = Hs[l], D = Ds[l], HD = H * D;

        dim3 gg((N + 63) / 64, HD / 64);
        gemm_wmma_bf16<<<gg, 128, 0, stream>>>(curx, W, hbuf, N, K, HD);

        attn_coef<<<N, HD, 0, stream>>>(hbuf, a_s, a_d, es, ed, H, D);

        layer_init<<<((size_t)N * HD + 255) / 256, 256, 0, stream>>>(
            obuf, mbuf, zbuf, N * HD, N * H);

        int tot = Ep * H;
        edge_max<<<(tot + 255) / 256, 256, 0, stream>>>(ei, E, Ep, H, es, ed, mbuf);
        edge_sum<<<(tot + 255) / 256, 256, 0, stream>>>(ei, E, Ep, H, es, ed, mbuf, zbuf);

        aggregate<<<Ep, HD, 0, stream>>>(ei, E, H, D, es, ed, mbuf, zbuf, hbuf, obuf);

        bn_stats<<<HD, 256, 0, stream>>>(obuf, N, HD, mu, inv);
        bn_apply_relu<<<((size_t)N * HD + 255) / 256, 256, 0, stream>>>(
            obuf, mu, inv, gamma, beta, xbuf, N * HD, HD);

        curx = xbuf;
    }

    pool_init<<<(NGROUP * FOUT + 255) / 256, 256, 0, stream>>>(fsum, cnt);
    pool_accum<<<((size_t)N * FOUT + 255) / 256, 256, 0, stream>>>(xbuf, batch, fsum, cnt, N);

    heads_kernel<<<NGROUP, 128, 0, stream>>>(
        fsum, cnt,
        (const float*)d_in[21], (const float*)d_in[22],
        (const float*)d_in[23], (const float*)d_in[24],
        (const float*)d_in[25], (const float*)d_in[26],
        (float*)d_out);
}
